// VRPPointerNetwork_34780645163055
// MI455X (gfx1250) — matrix operand order repaired and checked by reading, not compiled, not run on hardware
//
#include <hip/hip_runtime.h>
#include <hip/hip_bf16.h>
#include <math.h>

// ---------------------------------------------------------------------------
// VRP pointer network for MI455X (gfx1250, wave32, WMMA).
// B = N = H = 128, STEPS = 100, CAP = 20.
//
// Algebraic restructuring (exact):
//   x_t = enc_proj + (h@W2^T + b2)  (broadcast over N)
//   => x_t@Wa^T  = EA + hA,  EA = enc_proj@Wa^T + Wa_b   (precomputed once)
//   => x_t@Wwt^T = EW + hW,  EW = enc_proj@Wwt^T + Wwt_b (precomputed once)
//   => c_t = a_t@enc_proj + hw2     (softmax weights sum to 1)
//
// All GEMMs use v_wmma_f32_16x16x32_f16. Weights are pre-packed once into the
// exact WMMA B-operand register layout (f16), and the recurrent activations
// (x_in, h, hw2) are produced directly in packed A-operand layout, so the
// latency-critical per-step GEMMs do one 32-byte vector load per operand per
// lane and zero cvt work.
// ---------------------------------------------------------------------------

#define Bsz 128
#define Nn  128
#define Hd  128
#define STEPS 100
#define CAPF 20.0f

typedef __attribute__((ext_vector_type(16))) _Float16 v16h;
typedef __attribute__((ext_vector_type(8)))  float    v8f;

__device__ __forceinline__ float sigmoidf(float x) { return 1.0f / (1.0f + expf(-x)); }

// ---- WMMA operand helpers (CDNA5 ISA 7.12.2 register layouts) --------------
// A 16x32 f16: lanes 0-15 = rows, K in {0..7,16..23}; lanes 16-31 K in {8..15,24..31}
__device__ __forceinline__ v16h load_Af32(const float* __restrict__ A, int row0, int k0, int lda) {
  int lane = threadIdx.x & 31;
  int hl = lane >> 4, m = lane & 15;
  const float* p = A + (size_t)(row0 + m) * lda + k0;
  v16h a;
#pragma unroll
  for (int e = 0; e < 16; ++e) {
    int k = e + 8 * ((e >> 3) + hl);
    a[e] = (_Float16)p[k];
  }
  return a;
}

// B 32x16 f16 for X @ W^T: B(k,n) = W[col0+n][k0+k], W row-major [Nout,ldw].
// lanes 0-15 hold K in {0..7,16..23}, lanes 16-31 K in {8..15,24..31}, n = lane&15.
__device__ __forceinline__ v16h load_BtF32(const float* __restrict__ W, int col0, int k0, int ldw) {
  int lane = threadIdx.x & 31;
  int hl = lane >> 4, n = lane & 15;
  const float* p = W + (size_t)(col0 + n) * ldw + k0;
  v16h b;
#pragma unroll
  for (int e = 0; e < 16; ++e) b[e] = (_Float16)p[e + 8 * ((e >> 3) + hl)];
  return b;
}

// Packed operand fetch: one 32-byte vector load per lane (-> 2x global_load_b128)
__device__ __forceinline__ v16h load_pk(const _Float16* __restrict__ P, int tile) {
  return reinterpret_cast<const v16h*>(P)[(size_t)tile * 32 + (threadIdx.x & 31)];
}

// Inverse A-layout map: half index (within packed A buffer) holding element
// (row, k) of a [rows x 32*nkt] matrix stored as 16-row x 32-K tiles.
__device__ __forceinline__ size_t a_pack_idx(int row, int k, int nkt) {
  int rt = row >> 4, m = row & 15;
  int kt = k >> 5, kk = k & 31;
  int hl = (kk >> 3) & 1;
  int e  = (kk & 7) + 8 * (kk >> 4);
  int lane = m + 16 * hl;
  return ((size_t)(rt * nkt + kt) * 32 + lane) * 16 + e;
}

__device__ __forceinline__ v8f wmma_f16(v16h a, v16h b, v8f c) {
  return __builtin_amdgcn_wmma_f32_16x16x32_f16(false, a, false, b, (short)0, c, false, false);
}

// ---------------------------------------------------------------------------
// One-time kernels
// ---------------------------------------------------------------------------

// Pack W [Nout, ldw] into WMMA B-operand layout, tiles (ct, kt), f16.
__global__ __launch_bounds__(32) void pack_weightB(
    const float* __restrict__ W, int ldw, int nkt, _Float16* __restrict__ out)
{
  int ct = blockIdx.x, kt = blockIdx.y;
  v16h b = load_BtF32(W, ct * 16, kt * 32, ldw);
  reinterpret_cast<v16h*>(out)[(size_t)(ct * nkt + kt) * 32 + (threadIdx.x & 31)] = b;
}

// static / dyn embeddings (rank-2 inputs -> pointwise), demand copy, static^T out
__global__ __launch_bounds__(256) void encode_kernel(
    const float* __restrict__ loc, const float* __restrict__ demand,
    const float* __restrict__ Ws_w, const float* __restrict__ Ws_b,
    const float* __restrict__ Wd_w, const float* __restrict__ Wd_b,
    float* __restrict__ staticF, float* __restrict__ dyn,
    float* __restrict__ dem, float* __restrict__ outT)
{
  int idx = blockIdx.x * 256 + threadIdx.x;       // [B*N*H]
  int h = idx & 127, n = (idx >> 7) & 127, b = idx >> 14;
  float lx = loc[(size_t)(b * Nn + n) * 2 + 0];
  float ly = loc[(size_t)(b * Nn + n) * 2 + 1];
  float s = lx * Ws_w[h * 2 + 0] + ly * Ws_w[h * 2 + 1] + Ws_b[h];
  staticF[idx] = s;
  outT[((size_t)b * Hd + h) * Nn + n] = s;        // static transposed output
  float dm = demand[b * Nn + n];
  dyn[idx] = dm * Wd_w[h * 2 + 0] + (CAPF - dm) * Wd_w[h * 2 + 1] + Wd_b[h];
  if (h == 0) dem[b * Nn + n] = dm;
}

// depot embedding -> packed x_in, zero packed h / f32 c, load = CAP
__global__ __launch_bounds__(128) void init_state(
    const float* __restrict__ depot, const float* __restrict__ Ws_w,
    const float* __restrict__ Ws_b,
    _Float16* __restrict__ xinP, _Float16* __restrict__ hP,
    float* __restrict__ c, float* __restrict__ loadv)
{
  int b = blockIdx.x, t = threadIdx.x;
  float dx = depot[b * 2], dy = depot[b * 2 + 1];
  float v = dx * Ws_w[t * 2] + dy * Ws_w[t * 2 + 1] + Ws_b[t];
  size_t pi = a_pack_idx(b, t, Hd / 32);
  xinP[pi] = (_Float16)v;
  hP[pi] = (_Float16)0.0f;
  c[b * Hd + t] = 0.f;
  if (t == 0) loadv[b] = CAPF;
}

// Generic C = [Cin +] A(f32) @ Wpacked^T [+ bias], one 16x16 tile per wave32 block.
__global__ __launch_bounds__(32) void gemm_xwT(
    const float* __restrict__ A, const _Float16* __restrict__ Wp,
    const float* __restrict__ bias, const float* __restrict__ Cin,
    float* __restrict__ C, int N, int K)
{
  int tm = blockIdx.x * 16, tn = blockIdx.y * 16;
  int lane = threadIdx.x & 31, hl = lane >> 4, nn = lane & 15;
  int nkt = K / 32;
  v8f acc = {};
  if (Cin) {
#pragma unroll
    for (int r = 0; r < 8; ++r) acc[r] = Cin[(size_t)(tm + r + 8 * hl) * N + tn + nn];
  }
  for (int kt = 0; kt < nkt; ++kt)
    acc = wmma_f16(load_Af32(A, tm, kt * 32, K), load_pk(Wp, blockIdx.y * nkt + kt), acc);
  float bb = bias ? bias[tn + nn] : 0.0f;
#pragma unroll
  for (int r = 0; r < 8; ++r) C[(size_t)(tm + r + 8 * hl) * N + tn + nn] = acc[r] + bb;
}

// ---------------------------------------------------------------------------
// Per-step kernels
// ---------------------------------------------------------------------------

// Fused recurrent step (one 8-wave workgroup), all operands in packed layout:
//   gates = x@Wih^T + h@Whh^T + b  -> LSTM cell -> h (packed), c
//   hw2 = h@W2^T + W2_b (f32 + packed) ; hA = hw2@Wa^T ; hW = hw2@Wwt^T
__global__ __launch_bounds__(256) void step_recur(
    const _Float16* __restrict__ xinP, _Float16* __restrict__ hP,
    _Float16* __restrict__ hw2P, float* __restrict__ cbuf,
    float* __restrict__ gates,
    const _Float16* __restrict__ WihP, const _Float16* __restrict__ WhhP,
    const float* __restrict__ bih, const float* __restrict__ bhh,
    const _Float16* __restrict__ W2P, const float* __restrict__ W2b,
    const _Float16* __restrict__ WaP, const _Float16* __restrict__ WwtP,
    float* __restrict__ hw2, float* __restrict__ hA, float* __restrict__ hW)
{
  const int w = threadIdx.x >> 5;
  const int lane = threadIdx.x & 31, hl = lane >> 4, nn = lane & 15;
  const int nkt = Hd / 32;   // 4

  // Phase 1: gates [128 x 512]; wave w owns 4 column tiles
  for (int rt = 0; rt < 8; ++rt) {
    for (int q = 0; q < 4; ++q) {
      int ct = w * 4 + q;
      v8f acc = {};
#pragma unroll
      for (int kt = 0; kt < nkt; ++kt)
        acc = wmma_f16(load_pk(xinP, rt * nkt + kt), load_pk(WihP, ct * nkt + kt), acc);
#pragma unroll
      for (int kt = 0; kt < nkt; ++kt)
        acc = wmma_f16(load_pk(hP, rt * nkt + kt), load_pk(WhhP, ct * nkt + kt), acc);
      int col = ct * 16 + nn;
      float bb = bih[col] + bhh[col];
#pragma unroll
      for (int r = 0; r < 8; ++r) gates[(size_t)(rt * 16 + r + 8 * hl) * 512 + col] = acc[r] + bb;
    }
  }
  __threadfence(); __syncthreads();

  // Phase 2: LSTM cell (torch gate order i,f,g,o); h written in packed A layout
  for (int i = threadIdx.x; i < Bsz * Hd; i += 256) {
    int row = i >> 7, j = i & 127;
    const float* g = gates + (size_t)row * 512;
    float gi = g[j], gf = g[128 + j], gg = g[256 + j], go = g[384 + j];
    float cn = sigmoidf(gf) * cbuf[i] + sigmoidf(gi) * tanhf(gg);
    cbuf[i] = cn;
    hP[a_pack_idx(row, j, nkt)] = (_Float16)(sigmoidf(go) * tanhf(cn));
  }
  __threadfence(); __syncthreads();

  // Phase 3: hw2 = h@W2^T + W2_b  [128 x 128]; wave w owns column tile w
  for (int rt = 0; rt < 8; ++rt) {
    v8f acc = {};
#pragma unroll
    for (int kt = 0; kt < nkt; ++kt)
      acc = wmma_f16(load_pk(hP, rt * nkt + kt), load_pk(W2P, w * nkt + kt), acc);
    int col = w * 16 + nn;
    float bb = W2b[col];
#pragma unroll
    for (int r = 0; r < 8; ++r) {
      int row = rt * 16 + r + 8 * hl;
      float v = acc[r] + bb;
      hw2[(size_t)row * Hd + col] = v;                  // f32 for step_attn c_t
      hw2P[a_pack_idx(row, col, nkt)] = (_Float16)v;    // packed A for phase 4
    }
  }
  __threadfence(); __syncthreads();

  // Phase 4a: hA = hw2@Wa^T (no bias; Wa_b baked into EA)  [128 x 32]
  for (int q = 0; q < 2; ++q) {
    int ti = w * 2 + q, rt = ti >> 1, ct = ti & 1;
    v8f acc = {};
#pragma unroll
    for (int kt = 0; kt < nkt; ++kt)
      acc = wmma_f16(load_pk(hw2P, rt * nkt + kt), load_pk(WaP, ct * nkt + kt), acc);
#pragma unroll
    for (int r = 0; r < 8; ++r) hA[(size_t)(rt * 16 + r + 8 * hl) * 32 + ct * 16 + nn] = acc[r];
  }
  // Phase 4b: hW = hw2@Wwt^T (no bias; Wwt_b baked into EW) [128 x 128]
  for (int rt = 0; rt < 8; ++rt) {
    v8f acc = {};
#pragma unroll
    for (int kt = 0; kt < nkt; ++kt)
      acc = wmma_f16(load_pk(hw2P, rt * nkt + kt), load_pk(WwtP, w * nkt + kt), acc);
#pragma unroll
    for (int r = 0; r < 8; ++r) hW[(size_t)(rt * 16 + r + 8 * hl) * Hd + w * 16 + nn] = acc[r];
  }
}

// Attention: u_t -> softmax over N -> c_t = a_t@enc_proj + hw2   (1 block / batch)
__global__ __launch_bounds__(128) void step_attn(
    const float* __restrict__ EA, const float* __restrict__ hA,
    const float* __restrict__ Vaw, const float* __restrict__ Vab,
    const float* __restrict__ enc, const float* __restrict__ hw2,
    float* __restrict__ c_t)
{
  int b = blockIdx.x, tx = threadIdx.x;
  __shared__ float sm[Nn];
  const float* ea = EA + ((size_t)b * Nn + tx) * 32;
  const float* ha = hA + (size_t)b * 32;
  float u = Vab[0];
#pragma unroll 8
  for (int j = 0; j < 32; ++j) u += Vaw[j] * tanhf(ea[j] + ha[j]);
  sm[tx] = u;
  __syncthreads();
  float mx = -3.4e38f;
  for (int i = 0; i < Nn; ++i) mx = fmaxf(mx, sm[i]);
  float ssum = 0.f;
  for (int i = 0; i < Nn; ++i) ssum += expf(sm[i] - mx);
  float an = expf(sm[tx] - mx) / ssum;
  __syncthreads();
  sm[tx] = an;
  __syncthreads();
  // c_t[b, h=tx]
  float acc = hw2[(size_t)b * Hd + tx];
  const float* ep = enc + (size_t)b * Nn * Hd + tx;
  for (int i = 0; i < Nn; ++i) acc += sm[i] * ep[(size_t)i * Hd];
  c_t[(size_t)b * Hd + tx] = acc;
}

// Logits, log-softmax, masking, argmax, state update, outputs (1 block / batch)
__global__ __launch_bounds__(128) void step_logits(
    const float* __restrict__ EW, const float* __restrict__ hW,
    const float* __restrict__ c_t,
    const float* __restrict__ Wctw, const float* __restrict__ Wctb,
    const float* __restrict__ Vcw, const float* __restrict__ Vcb,
    float* __restrict__ dem, float* __restrict__ loadv,
    const float* __restrict__ staticF, _Float16* __restrict__ xinP,
    float* __restrict__ out_ptr, float* __restrict__ out_lp, int t)
{
  int b = blockIdx.x, tx = threadIdx.x;
  __shared__ float ctS[Hd], cwS[Hd], hwS[Hd], lp[Nn];
  __shared__ int ptr_s;
  ctS[tx] = c_t[(size_t)b * Hd + tx];
  hwS[tx] = hW[(size_t)b * Hd + tx];
  __syncthreads();
  // cW[h] = Wct_b[h] + c_t . Wct_w[h,:]
  {
    float acc = Wctb[tx];
    const float* wr = Wctw + (size_t)tx * Hd;
    for (int k = 0; k < Hd; ++k) acc += ctS[k] * wr[k];
    cwS[tx] = acc;
  }
  __syncthreads();
  // logit(n=tx) = Vc . tanh(EW + hW + cW) + Vc_b
  {
    const float* ew = EW + ((size_t)b * Nn + tx) * Hd;
    float s = Vcb[0];
    for (int h = 0; h < Hd; ++h) s += Vcw[h] * tanhf(ew[h] + hwS[h] + cwS[h]);
    lp[tx] = s;
  }
  __syncthreads();
  float mx = -3.4e38f;
  for (int i = 0; i < Nn; ++i) mx = fmaxf(mx, lp[i]);
  float se = 0.f;
  for (int i = 0; i < Nn; ++i) se += expf(lp[i] - mx);
  float myl = lp[tx] - mx - logf(se);
  // masks use pre-update carry state (reference semantics)
  float ld = loadv[b];
  if (tx >= 1 && (dem[b * Nn + tx] == 0.0f || ld == 0.0f)) myl = -__builtin_inff();
  __syncthreads();
  lp[tx] = myl;
  __syncthreads();
  if (tx == 0) {
    int p = 0; float best = lp[0];
    for (int i = 1; i < Nn; ++i) if (lp[i] > best) { best = lp[i]; p = i; }
    ptr_s = p;
    float dsel = dem[b * Nn + p];
    float nl = (p == 0) ? CAPF : (ld - dsel);      // load updated first
    loadv[b] = nl;
    dem[b * Nn + p] = (p == 0) ? dsel : (dsel - nl);
    out_ptr[b * STEPS + t] = (float)p;
  }
  __syncthreads();
  int p = ptr_s;
  // next LSTM input, written directly in packed WMMA A layout
  xinP[a_pack_idx(b, tx, Hd / 32)] = (_Float16)staticF[((size_t)b * Nn + p) * Hd + tx];
  out_lp[((size_t)b * STEPS + t) * Nn + tx] = myl;
}

// ---------------------------------------------------------------------------
// Host launcher
// ---------------------------------------------------------------------------
extern "C" void kernel_launch(void* const* d_in, const int* in_sizes, int n_in,
                              void* d_out, int out_size, void* d_ws, size_t ws_size,
                              hipStream_t stream) {
  (void)in_sizes; (void)n_in; (void)out_size; (void)ws_size;
  const float* location = (const float*)d_in[0];
  const float* demand   = (const float*)d_in[1];
  const float* depot    = (const float*)d_in[2];
  const float* Ws_w = (const float*)d_in[3];  const float* Ws_b = (const float*)d_in[4];
  const float* Wd_w = (const float*)d_in[5];  const float* Wd_b = (const float*)d_in[6];
  const float* Wih  = (const float*)d_in[7];  const float* Whh  = (const float*)d_in[8];
  const float* bih  = (const float*)d_in[9];  const float* bhh  = (const float*)d_in[10];
  const float* W1_w = (const float*)d_in[11]; const float* W1_b = (const float*)d_in[12];
  const float* W2_w = (const float*)d_in[13]; const float* W2_b = (const float*)d_in[14];
  const float* Wwt_w = (const float*)d_in[15]; const float* Wwt_b = (const float*)d_in[16];
  const float* Wct_w = (const float*)d_in[17]; const float* Wct_b = (const float*)d_in[18];
  const float* Wa_w = (const float*)d_in[19]; const float* Wa_b = (const float*)d_in[20];
  const float* Va_w = (const float*)d_in[21]; const float* Va_b = (const float*)d_in[22];
  const float* Vc_w = (const float*)d_in[23]; const float* Vc_b = (const float*)d_in[24];

  const size_t BNH = (size_t)Bsz * Nn * Hd;   // 2,097,152
  float* ws = (float*)d_ws;
  float* staticF = ws;                              // [B,N,H]
  float* enc     = staticF + BNH;                   // [B,N,H] enc_proj
  float* EW      = enc + BNH;                       // [B,N,H] (dyn first, then EW)
  float* EA      = EW + BNH;                        // [B,N,32]
  float* gates   = EA + (size_t)Bsz * Nn * 32;      // [B,512]
  float* cbuf    = gates + (size_t)Bsz * 512;
  float* hw2     = cbuf + (size_t)Bsz * Hd;
  float* hAv     = hw2  + (size_t)Bsz * Hd;         // [B,32]
  float* hWv     = hAv  + (size_t)Bsz * 32;         // [B,H]
  float* ctv     = hWv  + (size_t)Bsz * Hd;         // [B,H]
  float* dem     = ctv  + (size_t)Bsz * Hd;         // [B,N] mutable demand copy
  float* loadv   = dem  + (size_t)Bsz * Nn;         // [B]
  float* pk      = loadv + 128;                     // packed f16 region (32B aligned)
  _Float16* xinP = (_Float16*)pk;                   // [B,H] packed A  (16384 h)
  _Float16* hP   = xinP + (size_t)Bsz * Hd;
  _Float16* hw2P = hP   + (size_t)Bsz * Hd;
  _Float16* WihP = hw2P + (size_t)Bsz * Hd;         // 512x128 packed B (65536 h)
  _Float16* WhhP = WihP + (size_t)512 * Hd;
  _Float16* W2P  = WhhP + (size_t)512 * Hd;         // 128x128
  _Float16* WwtP = W2P  + (size_t)Hd * Hd;
  _Float16* WaP  = WwtP + (size_t)Hd * Hd;          // 32x128
  _Float16* W1sP = WaP  + (size_t)32 * Hd;          // 128x128 (W1 static half)
  _Float16* W1dP = W1sP + (size_t)Hd * Hd;          // 128x128 (W1 dynamic half)

  float* out0 = (float*)d_out;                      // ptrs [B,100]
  float* out1 = out0 + (size_t)Bsz * STEPS;         // log-probs [B,100,N]
  float* out2 = out1 + (size_t)Bsz * STEPS * Nn;    // static^T [B,H,N]

  // ---- one-time: weight packing into WMMA B-operand layout ----
  pack_weightB<<<dim3(32, 4), 32, 0, stream>>>(Wih,        Hd,     4, WihP);
  pack_weightB<<<dim3(32, 4), 32, 0, stream>>>(Whh,        Hd,     4, WhhP);
  pack_weightB<<<dim3(8, 4),  32, 0, stream>>>(W2_w,       Hd,     4, W2P);
  pack_weightB<<<dim3(8, 4),  32, 0, stream>>>(Wwt_w,      Hd,     4, WwtP);
  pack_weightB<<<dim3(2, 4),  32, 0, stream>>>(Wa_w,       Hd,     4, WaP);
  pack_weightB<<<dim3(8, 4),  32, 0, stream>>>(W1_w,       2 * Hd, 4, W1sP);
  pack_weightB<<<dim3(8, 4),  32, 0, stream>>>(W1_w + Hd,  2 * Hd, 4, W1dP);

  // ---- one-time: encoder + step-invariant projections ----
  encode_kernel<<<(int)(BNH / 256), 256, 0, stream>>>(
      location, demand, Ws_w, Ws_b, Wd_w, Wd_b, staticF, /*dyn=*/EW, dem, out2);
  init_state<<<Bsz, Hd, 0, stream>>>(depot, Ws_w, Ws_b, xinP, hP, cbuf, loadv);

  dim3 gBig(Bsz * Nn / 16, Hd / 16);    // 1024 x 8
  // enc_proj = static @ W1s^T + dyn @ W1d^T + W1_b
  gemm_xwT<<<gBig, 32, 0, stream>>>(staticF,   W1sP, W1_b,  nullptr, enc, Hd, Hd);
  gemm_xwT<<<gBig, 32, 0, stream>>>(/*dyn*/EW, W1dP, nullptr, enc,   enc, Hd, Hd);
  // EA = enc @ Wa^T + Wa_b ; EW = enc @ Wwt^T + Wwt_b (overwrites dyn)
  dim3 gEA(Bsz * Nn / 16, 32 / 16);
  gemm_xwT<<<gEA,  32, 0, stream>>>(enc, WaP,  Wa_b,  nullptr, EA, 32, Hd);
  gemm_xwT<<<gBig, 32, 0, stream>>>(enc, WwtP, Wwt_b, nullptr, EW, Hd, Hd);

  // ---- sequential decode: 100 steps, 3 launches each ----
  for (int t = 0; t < STEPS; ++t) {
    step_recur<<<1, 256, 0, stream>>>(xinP, hP, hw2P, cbuf, gates,
                                      WihP, WhhP, bih, bhh, W2P, W2_b, WaP, WwtP,
                                      hw2, hAv, hWv);
    step_attn<<<Bsz, Nn, 0, stream>>>(EA, hAv, Va_w, Va_b, enc, hw2, ctv);
    step_logits<<<Bsz, Nn, 0, stream>>>(EW, hWv, ctv, Wct_w, Wct_b, Vc_w, Vc_b,
                                        dem, loadv, staticF, xinP, out0, out1, t);
  }
}
